// Quantizer_9990093931135
// MI455X (gfx1250) — matrix operand order repaired and checked by reading, not compiled, not run on hardware
//
#include <hip/hip_runtime.h>
#include <hip/hip_bf16.h>

// ---------------------------------------------------------------------------
// VQ-VAE quantizer for MI455X (gfx1250, wave32).
//
// Shapes (fixed by reference): input [32,64,64,64] f32 -> flat [131072, 64]
//   embed [64,512], cluster_size [512], embed_avg [64,512]
//
// dist = ||x||^2 - 2 x.e + ||e||^2 ; argmin over 512 codes. ||x||^2 is
// row-constant -> irrelevant for argmin. GEMM done with
// v_wmma_f32_16x16x32_bf16 using a hi/lo bf16 split of both operands
// (3 products, xl*el dropped) for ~f32-grade argmin at bf16 WMMA rates.
// ---------------------------------------------------------------------------

typedef float  v8f   __attribute__((ext_vector_type(8)));
typedef __bf16 v16bf __attribute__((ext_vector_type(16)));
typedef float  f4v   __attribute__((ext_vector_type(4)));

#define DECAY   0.99f
#define ONE_MD  0.01f
#define EPS_    1e-5f

static __device__ __forceinline__ v8f wmma_bf16(v16bf a, v16bf b, v8f c) {
  return __builtin_amdgcn_wmma_f32_16x16x32_bf16(
      /*neg_a=*/false, a, /*neg_b=*/false, b,
      /*c_mod=*/(short)0, c, /*reuse_a=*/false, /*reuse_b=*/false);
}

// ---------------------------------------------------------------------------
// Prep 1: embedT (coalesced gather source), column norms, zero accumulators.
// 32768 threads.
// ---------------------------------------------------------------------------
__global__ void vq_prep(const float* __restrict__ embed,
                        float* __restrict__ embedT,
                        float* __restrict__ cnorm,
                        float* __restrict__ cnt_acc,
                        float* __restrict__ esum_acc,
                        float* __restrict__ sumsq_acc) {
  int t = blockIdx.x * 256 + threadIdx.x;          // 0..32767
  int n = t >> 6, k = t & 63;
  embedT[t]   = embed[k * 512 + n];                // embedT[n][k]
  esum_acc[t] = 0.0f;
  if (t < 512) {
    float s = 0.0f;
    for (int kk = 0; kk < 64; ++kk) {
      float e = embed[kk * 512 + t];
      s += e * e;
    }
    cnorm[t]   = s;
    cnt_acc[t] = 0.0f;
  }
  if (t == 0) sumsq_acc[0] = 0.0f;
}

// ---------------------------------------------------------------------------
// Prep 2: build WMMA B-fragments (hi/lo bf16) for all 32 column tiles x 2
// K-chunks. Slot s = (j*2+c)*32 + lane holds the v16bf one lane consumes.
// B k order used here (contiguous, same as A): lane L -> column N = L%16; lanes 0-15 hold
// K = base..base+15 with element i = K base+i; lanes 16-31 hold +16.
// 2048 threads (one per slot).
// ---------------------------------------------------------------------------
__global__ void vq_fragprep(const float* __restrict__ embed,
                            __bf16* __restrict__ fhi,
                            __bf16* __restrict__ flo) {
  int t = blockIdx.x * 256 + threadIdx.x;          // 0..2047
  int lane = t & 31;
  int c    = (t >> 5) & 1;
  int j    = t >> 6;
  int n     = j * 16 + (lane & 15);
  int kbase = c * 32 + (lane >> 4) * 16;
#pragma unroll
  for (int i = 0; i < 16; ++i) {
    float  e  = embed[(kbase + i) * 512 + n];
    __bf16 hi = (__bf16)e;
    __bf16 lo = (__bf16)(e - (float)hi);
    fhi[t * 16 + i] = hi;
    flo[t * 16 + i] = lo;
  }
}

// ---------------------------------------------------------------------------
// Main kernel: 1024 blocks x 256 threads (8 waves). Each wave owns 16 rows.
// ---------------------------------------------------------------------------
__global__ __launch_bounds__(256) void vq_main(
    const float*  __restrict__ inp,      // [131072, 64]
    const float*  __restrict__ cnorm,    // [512]
    const __bf16* __restrict__ fhi,      // B frags hi
    const __bf16* __restrict__ flo,      // B frags lo
    const float*  __restrict__ embedT,   // [512, 64]
    float* __restrict__ q_out,           // [131072, 64]
    float* __restrict__ inds_out,        // [131072] (as float)
    float* __restrict__ cnt_acc,         // [512]
    float* __restrict__ esum_acc,        // [512, 64]  (transposed embed_sum)
    float* __restrict__ sumsq_acc) {     // [1]
  const int lane    = threadIdx.x & 31;
  const int wave    = threadIdx.x >> 5;
  const int rowBase = blockIdx.x * 128 + wave * 16;
  const int m       = lane & 15;   // row within tile (A) / column (B,C)
  const int h       = lane >> 4;   // half-wave

  // ----- Build A fragments (16 rows x K-chunk of 32) hi/lo bf16 -----
  // A k order used here (contiguous, same as B): lane L -> row M = L%16.
  //   elements 0..7  : K = c*32 + h*16 +     (0..7)
  //   elements 8..15 : K = c*32 + h*16 + 8 + (0..7)
  const float* rowp = inp + (size_t)(rowBase + m) * 64;
  v16bf ahi[2], alo[2];
#pragma unroll
  for (int c = 0; c < 2; ++c) {
#pragma unroll
    for (int run = 0; run < 2; ++run) {
      const f4v* p = (const f4v*)(rowp + c * 32 + h * 16 + run * 8);
      f4v r0 = p[0];
      f4v r1 = p[1];
#pragma unroll
      for (int i = 0; i < 4; ++i) {
        float  x0 = r0[i];
        __bf16 h0 = (__bf16)x0;
        ahi[c][run * 8 + i] = h0;
        alo[c][run * 8 + i] = (__bf16)(x0 - (float)h0);
        float  x1 = r1[i];
        __bf16 h1 = (__bf16)x1;
        ahi[c][run * 8 + 4 + i] = h1;
        alo[c][run * 8 + 4 + i] = (__bf16)(x1 - (float)h1);
      }
    }
  }

  // ----- Sweep 32 column tiles of 16 codes each -----
  float bestd[8];
  int   besti[8];
#pragma unroll
  for (int v = 0; v < 8; ++v) { bestd[v] = __builtin_inff(); besti[v] = 0; }

  const v16bf* bhi = (const v16bf*)fhi;
  const v16bf* blo = (const v16bf*)flo;

  for (int j = 0; j < 32; ++j) {
    int slot0 = (j * 2 + 0) * 32 + lane;
    v16bf bh0 = bhi[slot0];
    v16bf bh1 = bhi[slot0 + 32];
    v16bf bl0 = blo[slot0];
    v16bf bl1 = blo[slot0 + 32];

    v8f acc = {};
    acc = wmma_bf16(ahi[0], bh0, acc);
    acc = wmma_bf16(ahi[1], bh1, acc);
    acc = wmma_bf16(alo[0], bh0, acc);
    acc = wmma_bf16(alo[1], bh1, acc);
    acc = wmma_bf16(ahi[0], bl0, acc);
    acc = wmma_bf16(ahi[1], bl1, acc);

    float cn = cnorm[j * 16 + m];
    int   nidx = j * 16 + m;
#pragma unroll
    for (int v = 0; v < 8; ++v) {
      float d = cn - 2.0f * acc[v];    // ||e||^2 - 2 x.e
      if (d < bestd[v]) { bestd[v] = d; besti[v] = nidx; }
    }
  }

  // ----- argmin reduce across the 16 lanes of each half-wave -----
  // element v of lane L corresponds to row M = v + 8*(L/16); its 512
  // candidates live across the 16 lanes of that half (N = lane%16 + 16*j).
#pragma unroll
  for (int off = 1; off < 16; off <<= 1) {
#pragma unroll
    for (int v = 0; v < 8; ++v) {
      float od = __shfl_xor(bestd[v], off, 32);
      int   oi = __shfl_xor(besti[v], off, 32);
      if (od < bestd[v] || (od == bestd[v] && oi < besti[v])) {
        bestd[v] = od;
        besti[v] = oi;
      }
    }
  }

  // ----- emit indices, quantized gather, segment sums, loss partial -----
  float ss = 0.0f;
#pragma unroll
  for (int v = 0; v < 8; ++v) {
    int r   = rowBase + v + 8 * h;
    int idx = besti[v];               // identical across the 16-lane half
    if (m == 0) {
      inds_out[r] = (float)idx;
      atomicAdd(cnt_acc + idx, 1.0f);
    }
    int col = m * 4;
    f4v q = *(const f4v*)(embedT + (size_t)idx * 64 + col);
    f4v x = *(const f4v*)(inp + (size_t)r * 64 + col);
    *(f4v*)(q_out + (size_t)r * 64 + col) = q;
#pragma unroll
    for (int i = 0; i < 4; ++i) {
      float dd = q[i] - x[i];
      ss += dd * dd;
      atomicAdd(esum_acc + (size_t)idx * 64 + col + i, x[i]);
    }
  }

  // loss partial: reduce over wave, one atomic per wave
#pragma unroll
  for (int off = 16; off > 0; off >>= 1) ss += __shfl_xor(ss, off, 32);
  if (lane == 0) atomicAdd(sumsq_acc, ss);
}

// ---------------------------------------------------------------------------
// Finalize: EMA buffer math + loss. 1 block x 512 threads.
// ---------------------------------------------------------------------------
__global__ void vq_finalize(const float* __restrict__ cluster_size,
                            const float* __restrict__ embed_avg,
                            const float* __restrict__ cnt_acc,
                            const float* __restrict__ esum_acc,
                            const float* __restrict__ sumsq_acc,
                            float* __restrict__ out_loss,
                            float* __restrict__ out_new_embed,
                            float* __restrict__ out_new_cs,
                            float* __restrict__ out_new_ea) {
  __shared__ float red[512];
  int t = threadIdx.x;
  float ncs = cluster_size[t] * DECAY + ONE_MD * cnt_acc[t];
  out_new_cs[t] = ncs;
  red[t] = ncs;
  __syncthreads();
  for (int s = 256; s > 0; s >>= 1) {
    if (t < s) red[t] += red[t + s];
    __syncthreads();
  }
  float ntot = red[0];
  float cs   = (ncs + EPS_) / (ntot + 512.0f * EPS_) * ntot;
  float inv  = 1.0f / cs;
#pragma unroll 4
  for (int k = 0; k < 64; ++k) {
    float nea = embed_avg[k * 512 + t] * DECAY + ONE_MD * esum_acc[t * 64 + k];
    out_new_ea[k * 512 + t]    = nea;
    out_new_embed[k * 512 + t] = nea * inv;
  }
  if (t == 0) {
    // vq_loss = commitment*0.25 + embedding = 1.25 * MSE
    out_loss[0] = 1.25f * sumsq_acc[0] / 8388608.0f;
  }
}

// ---------------------------------------------------------------------------
extern "C" void kernel_launch(void* const* d_in, const int* in_sizes, int n_in,
                              void* d_out, int out_size, void* d_ws, size_t ws_size,
                              hipStream_t stream) {
  const float* inp          = (const float*)d_in[0];  // [32,64,64,64]
  const float* embed        = (const float*)d_in[1];  // [64,512]
  const float* cluster_size = (const float*)d_in[2];  // [512]
  const float* embed_avg    = (const float*)d_in[3];  // [64,512]

  float* out = (float*)d_out;
  float* o_q    = out;                 // 8388608
  float* o_loss = out + 8388608;       // 1
  float* o_inds = out + 8388609;       // 131072
  float* o_ne   = out + 8519681;       // 32768
  float* o_ncs  = out + 8552449;       // 512
  float* o_nea  = out + 8552961;       // 32768

  char* ws = (char*)d_ws;
  __bf16* fhi    = (__bf16*)(ws);               // 65536 B
  __bf16* flo    = (__bf16*)(ws + 65536);       // 65536 B
  float*  cnorm  = (float*)(ws + 131072);       // 2048 B
  float*  embedT = (float*)(ws + 133120);       // 131072 B
  float*  cntac  = (float*)(ws + 264192);       // 2048 B
  float*  esum   = (float*)(ws + 266240);       // 131072 B
  float*  sumsq  = (float*)(ws + 397312);       // 4 B

  vq_prep<<<128, 256, 0, stream>>>(embed, embedT, cnorm, cntac, esum, sumsq);
  vq_fragprep<<<8, 256, 0, stream>>>(embed, fhi, flo);
  vq_main<<<1024, 256, 0, stream>>>(inp, cnorm, fhi, flo, embedT,
                                    o_q, o_inds, cntac, esum, sumsq);
  vq_finalize<<<1, 512, 0, stream>>>(cluster_size, embed_avg, cntac, esum,
                                     sumsq, o_loss, o_ne, o_ncs, o_nea);
}
